// NonLocalBlock2D_23416161698526
// MI455X (gfx1250) — matrix operand order repaired and checked by reading, not compiled, not run on hardware
//
#include <hip/hip_runtime.h>
#include <hip/hip_bf16.h>

typedef __attribute__((ext_vector_type(16))) __bf16 v16bf;
typedef __attribute__((ext_vector_type(8)))  float  v8f;
typedef int v4i __attribute__((ext_vector_type(4)));
typedef unsigned short u16;
typedef unsigned int   u32;

#define BATCH 8
#define NQ    4096
#define NK    1024
#define MTOT  (BATCH * NQ)   // 32768
#define LDT   40             // padded LDS stride for 32-wide K tiles

// CDNA5 async global->LDS path (guarded: fall back to sync loads if absent)
#if defined(__has_builtin)
#if __has_builtin(__builtin_amdgcn_global_load_async_to_lds_b128) && \
    __has_builtin(__builtin_amdgcn_s_wait_asynccnt)
#define USE_ASYNC_LDS 1
#endif
#endif
#ifdef USE_ASYNC_LDS
typedef __attribute__((address_space(1))) v4i as1_v4i;  // global int4
typedef __attribute__((address_space(3))) v4i as3_v4i;  // LDS int4
#endif

// ---------- bf16 helpers ----------
__device__ __forceinline__ u16 f2bf(float f) {
  u32 u = __float_as_uint(f);
  u32 r = (u + 0x7FFFu + ((u >> 16) & 1u)) >> 16;  // round-to-nearest-even
  return (u16)r;
}

union FragBF { v16bf v; uint4 q[2]; u16 s[16]; };

// ---------- WMMA fragment loaders (per CDNA5 ISA 7.12.2 layouts) ----------
// A 16x32 bf16: lane = half*16+m ; e<8 -> K = e + half*8 ; e>=8 -> K = (e-8)+16+half*8
__device__ __forceinline__ v16bf load_a_frag(const u16* lds, int row0, int ldk, int k0) {
  int lane = threadIdx.x & 31;
  const u16* p = lds + (row0 + (lane & 15)) * ldk + k0 + ((lane >> 4) << 3);
  FragBF u;
  u.q[0] = *(const uint4*)(p);
  u.q[1] = *(const uint4*)(p + 16);
  return u.v;
}
// B 32x16 bf16, matrix stored row-major [N][K] (we compute A*B^T): lane n=L%16, e<8 -> K = e + half*8 ; e>=8 -> K = (e-8)+16+half*8
__device__ __forceinline__ v16bf load_bt_frag(const u16* lds, int n0, int ldk, int k0) {
  int lane = threadIdx.x & 31;
  const u16* p = lds + (n0 + (lane & 15)) * ldk + k0 + ((lane >> 4) << 3);
  FragBF u;
  u.q[0] = *(const uint4*)(p);
  u.q[1] = *(const uint4*)(p + 16);
  return u.v;
}

__device__ __forceinline__ v8f wmma_bf16(v16bf a, v16bf b, v8f c) {
  return __builtin_amdgcn_wmma_f32_16x16x32_bf16(false, a, false, b, (short)0, c, false, false);
}
__device__ __forceinline__ v8f vzero8() {
  v8f z;
#pragma unroll
  for (int e = 0; e < 8; ++e) z[e] = 0.f;
  return z;
}

// ---------- shared GEMM mainloop: C[128x128] += A[M,K](bf16,rm) * B[N,K](bf16,rm)^T ----------
struct Acc8 { v8f c[2][4]; };

__device__ __forceinline__ void gemm_mainloop(const u16* __restrict__ A, const u16* __restrict__ B,
                                              int K, int m_blk, int n_blk,
                                              u16* ldsA, u16* ldsB, Acc8& acc) {
  const int tid  = threadIdx.x;
  const int wave = tid >> 5;
  const int wm   = wave & 3, wn = wave >> 2;   // 4x2 waves -> 32x64 per wave
  const int lr   = tid >> 1;                   // 0..127 tile row
  const int lc   = (tid & 1) << 4;             // 0 / 16

#pragma unroll
  for (int i = 0; i < 2; ++i)
#pragma unroll
    for (int j = 0; j < 4; ++j) acc.c[i][j] = vzero8();

  for (int k0 = 0; k0 < K; k0 += 32) {
    uint4 av = *(const uint4*)(A + (size_t)(m_blk + lr) * K + k0 + lc);
    uint4 bv = *(const uint4*)(B + (size_t)(n_blk + lr) * K + k0 + lc);
    if (k0 + 32 < K) {
      __builtin_prefetch(A + (size_t)(m_blk + lr) * K + k0 + 32 + lc, 0, 1);
      __builtin_prefetch(B + (size_t)(n_blk + lr) * K + k0 + 32 + lc, 0, 1);
    }
    __syncthreads();
    *(uint4*)(ldsA + lr * LDT + lc) = av;
    *(uint4*)(ldsB + lr * LDT + lc) = bv;
    __syncthreads();
    v16bf a0 = load_a_frag(ldsA, wm * 32,      LDT, 0);
    v16bf a1 = load_a_frag(ldsA, wm * 32 + 16, LDT, 0);
#pragma unroll
    for (int j = 0; j < 4; ++j) {
      v16bf bj = load_bt_frag(ldsB, wn * 64 + j * 16, LDT, 0);
      acc.c[0][j] = wmma_bf16(a0, bj, acc.c[0][j]);
      acc.c[1][j] = wmma_bf16(a1, bj, acc.c[1][j]);
    }
  }
}

// ---------- kernels ----------
// x [B][256][4096] f32 -> xT [B][4096][256] bf16
__global__ __launch_bounds__(256) void k_transpose_bf16(const float* __restrict__ x, u16* __restrict__ xT) {
  __shared__ u16 t[32][72];
  int b = blockIdx.z, c0 = blockIdx.y * 32, q0 = blockIdx.x * 64;
  int tid = threadIdx.x;
  {
    int r = tid >> 3, qc = (tid & 7) * 8;
    const float* p = x + ((size_t)(b * 256 + c0 + r) << 12) + q0 + qc;
#pragma unroll
    for (int e = 0; e < 8; ++e) t[r][qc + e] = f2bf(p[e]);
  }
  __syncthreads();
  {
    int qr = tid >> 2, cc = (tid & 3) * 8;
    union { u16 s[8]; uint4 q; } tt;
#pragma unroll
    for (int e = 0; e < 8; ++e) tt.s[e] = t[cc + e][qr];
    *(uint4*)(xT + ((size_t)(b * 4096 + q0 + qr) << 8) + c0 + cc) = tt.q;
  }
}

__global__ void k_f32_to_bf16(const float* __restrict__ s, u16* __restrict__ d, int n) {
  int i = blockIdx.x * 256 + threadIdx.x;
  if (i < n) d[i] = f2bf(s[i]);
}

// conv1x1 -> f32 [M][N] (g / phi paths)
__global__ __launch_bounds__(256) void k_conv_f32(const u16* __restrict__ A, const u16* __restrict__ B,
                                                  const float* __restrict__ bias, float* __restrict__ out,
                                                  int K, int N) {
  __shared__ u16 ldsA[128 * LDT];
  __shared__ u16 ldsB[128 * LDT];
  Acc8 acc;
  int m_blk = blockIdx.y * 128, n_blk = blockIdx.x * 128;
  gemm_mainloop(A, B, K, m_blk, n_blk, ldsA, ldsB, acc);
  int lane = threadIdx.x & 31, wave = threadIdx.x >> 5;
  int wm = wave & 3, wn = wave >> 2, half = lane >> 4, l15 = lane & 15;
#pragma unroll
  for (int j = 0; j < 4; ++j) {
    int n = n_blk + wn * 64 + j * 16 + l15;
    float bv = bias[n];
#pragma unroll
    for (int i = 0; i < 2; ++i)
#pragma unroll
      for (int r = 0; r < 8; ++r) {
        int m = m_blk + wm * 32 + i * 16 + half * 8 + r;
        out[(size_t)m * N + n] = acc.c[i][j][r] + bv;
      }
  }
}

// conv1x1 theta -> fused row L2-norm -> bf16 [M][128]
__global__ __launch_bounds__(256) void k_conv_theta(const u16* __restrict__ A, const u16* __restrict__ B,
                                                    const float* __restrict__ bias, u16* __restrict__ out,
                                                    int K) {
  __shared__ u16 ldsA[128 * LDT];
  __shared__ u16 ldsB[128 * LDT];
  __shared__ float ssrow[128];
  Acc8 acc;
  int m_blk = blockIdx.y * 128;
  gemm_mainloop(A, B, K, m_blk, 0, ldsA, ldsB, acc);
  int tid = threadIdx.x, lane = tid & 31, wave = tid >> 5;
  int wm = wave & 3, wn = wave >> 2, half = lane >> 4, l15 = lane & 15;
  __syncthreads();
  if (tid < 128) ssrow[tid] = 0.f;
  __syncthreads();
#pragma unroll
  for (int j = 0; j < 4; ++j) {
    float bv = bias[wn * 64 + j * 16 + l15];
#pragma unroll
    for (int i = 0; i < 2; ++i)
#pragma unroll
      for (int r = 0; r < 8; ++r) {
        float v = acc.c[i][j][r] + bv;
        acc.c[i][j][r] = v;
        atomicAdd(&ssrow[wm * 32 + i * 16 + half * 8 + r], v * v);
      }
  }
  __syncthreads();
#pragma unroll
  for (int j = 0; j < 4; ++j) {
    int n = wn * 64 + j * 16 + l15;
#pragma unroll
    for (int i = 0; i < 2; ++i)
#pragma unroll
      for (int r = 0; r < 8; ++r) {
        int ml = wm * 32 + i * 16 + half * 8 + r;
        float sc = 1.f / fmaxf(sqrtf(ssrow[ml]), 1e-12f);
        out[(size_t)(m_blk + ml) * 128 + n] = f2bf(acc.c[i][j][r] * sc);
      }
  }
}

// 2x2 maxpool of gconv/phconv [B*4096][128] -> [B*1024][128]; phi gets per-row L2 norm
__global__ __launch_bounds__(128) void k_poolnorm(const float* __restrict__ gconv, const float* __restrict__ phconv,
                                                  u16* __restrict__ gpool, u16* __restrict__ phpool) {
  __shared__ float red[128];
  int bk = blockIdx.x;                 // b*1024 + kq
  int kq = bk & 1023, b = bk >> 10;
  int kh = kq >> 5, kw = kq & 31;
  int q00 = (kh * 2) * 64 + kw * 2;
  size_t base = ((size_t)(b * 4096 + q00)) * 128 + threadIdx.x;
  float gm = fmaxf(fmaxf(gconv[base], gconv[base + 128]),
                   fmaxf(gconv[base + 64 * 128], gconv[base + 65 * 128]));
  float pm = fmaxf(fmaxf(phconv[base], phconv[base + 128]),
                   fmaxf(phconv[base + 64 * 128], phconv[base + 65 * 128]));
  size_t obase = ((size_t)bk) * 128 + threadIdx.x;
  gpool[obase] = f2bf(gm);
  red[threadIdx.x] = pm * pm;
  __syncthreads();
  for (int st = 64; st > 0; st >>= 1) {
    if ((int)threadIdx.x < st) red[threadIdx.x] += red[threadIdx.x + st];
    __syncthreads();
  }
  float sc = 1.f / fmaxf(sqrtf(red[0]), 1e-12f);
  phpool[obase] = f2bf(pm * sc);
}

// fused attention: out[b,q,ic] = sum_k (th[q,:]·ph[k,:]) / Nk * g[k,ic]
__global__ __launch_bounds__(256) void k_attention(const u16* __restrict__ th, const u16* __restrict__ ph,
                                                   const u16* __restrict__ gp, u16* __restrict__ attn) {
  __shared__ u16 lds_th[128 * 136];   // th tile   [q][ic]
  __shared__ u16 lds_ph[32 * 136];    // ph chunk  [kq][ic]
  __shared__ u16 lds_gT[128 * LDT];   // g chunk transposed [ic][kq] -> fast B-frag loads
  __shared__ u16 lds_f[128 * LDT];    // f chunk   [q][kq] bf16
  int b = blockIdx.y, q0 = blockIdx.x * 128;
  int tid = threadIdx.x, lane = tid & 31, wave = tid >> 5;
  int wm = wave & 3, wn = wave >> 2, half = lane >> 4, l15 = lane & 15;
  {
    int r = tid >> 1, cc = (tid & 1) * 64;
    const u16* p = th + ((size_t)(b * 4096 + q0 + r) << 7) + cc;
#pragma unroll
    for (int e = 0; e < 8; ++e)
      *(uint4*)(lds_th + r * 136 + cc + e * 8) = *(const uint4*)(p + e * 8);
  }
  v8f acc[2][4];
#pragma unroll
  for (int i = 0; i < 2; ++i)
#pragma unroll
    for (int j = 0; j < 4; ++j) acc[i][j] = vzero8();
  const float invNk = 1.0f / (float)NK;

  for (int kq0 = 0; kq0 < NK; kq0 += 32) {
    int r = tid >> 3, cc = (tid & 7) * 16;
    const u16* pg = gp + ((size_t)(b * 1024 + kq0 + r) << 7) + cc;
    union { uint4 q[2]; u16 s[16]; } gg;
    gg.q[0] = *(const uint4*)(pg);
    gg.q[1] = *(const uint4*)(pg + 8);
    const u16* pp = ph + ((size_t)(b * 1024 + kq0 + r) << 7) + cc;
#ifndef USE_ASYNC_LDS
    uint4 pv0 = *(const uint4*)(pp), pv1 = *(const uint4*)(pp + 8);
#endif
    __syncthreads();                       // prior stage reads complete
#ifdef USE_ASYNC_LDS
    // CDNA5 async DMA: global -> LDS directly, tracked by ASYNCcnt
    __builtin_amdgcn_global_load_async_to_lds_b128(
        (as1_v4i*)(pp), (as3_v4i*)(lds_ph + r * 136 + cc), 0, 0);
    __builtin_amdgcn_global_load_async_to_lds_b128(
        (as1_v4i*)(pp + 8), (as3_v4i*)(lds_ph + r * 136 + cc + 8), 0, 0);
#else
    *(uint4*)(lds_ph + r * 136 + cc) = pv0;
    *(uint4*)(lds_ph + r * 136 + cc + 8) = pv1;
#endif
    // transpose g chunk into [ic][kq] so stage 2 uses contiguous B-frag loads
#pragma unroll
    for (int e = 0; e < 16; ++e) lds_gT[(cc + e) * LDT + r] = gg.s[e];
#ifdef USE_ASYNC_LDS
    __builtin_amdgcn_s_wait_asynccnt(0);
#endif
    __syncthreads();
    // stage 1: f[wave*16 .. +16][0..32] = th_tile x ph_chunk^T  (K = 128)
    v8f f0 = vzero8(), f1 = vzero8();
#pragma unroll
    for (int ks = 0; ks < 128; ks += 32) {
      v16bf a  = load_a_frag(lds_th, wave * 16, 136, ks);
      v16bf b0 = load_bt_frag(lds_ph, 0, 136, ks);
      v16bf b1 = load_bt_frag(lds_ph, 16, 136, ks);
      f0 = wmma_bf16(a, b0, f0);
      f1 = wmma_bf16(a, b1, f1);
    }
#pragma unroll
    for (int rr = 0; rr < 8; ++rr) {
      int m = wave * 16 + half * 8 + rr;
      lds_f[m * LDT + l15]      = f2bf(f0[rr] * invNk);
      lds_f[m * LDT + 16 + l15] = f2bf(f1[rr] * invNk);
    }
    __syncthreads();
    // stage 2: acc += f_chunk[128x32] x g_chunk[32x128]
    v16bf a0 = load_a_frag(lds_f, wm * 32,      LDT, 0);
    v16bf a1 = load_a_frag(lds_f, wm * 32 + 16, LDT, 0);
#pragma unroll
    for (int j = 0; j < 4; ++j) {
      v16bf bj = load_bt_frag(lds_gT, wn * 64 + j * 16, LDT, 0);
      acc[0][j] = wmma_bf16(a0, bj, acc[0][j]);
      acc[1][j] = wmma_bf16(a1, bj, acc[1][j]);
    }
  }
#pragma unroll
  for (int j = 0; j < 4; ++j) {
    int n = wn * 64 + j * 16 + l15;
#pragma unroll
    for (int i = 0; i < 2; ++i)
#pragma unroll
      for (int rr = 0; rr < 8; ++rr) {
        int m = q0 + wm * 32 + i * 16 + half * 8 + rr;
        attn[((size_t)(b * 4096 + m) << 7) + n] = f2bf(acc[i][j][rr]);
      }
  }
}

// final conv: Wy[b][c][q] = attn[m,:]·W_w[c,:] + W_b[c], stored transposed into d_out
__global__ __launch_bounds__(256) void k_conv_final(const u16* __restrict__ A, const u16* __restrict__ B,
                                                    const float* __restrict__ bias, float* __restrict__ outWy,
                                                    int K) {
  __shared__ u16 ldsA[128 * LDT];
  __shared__ u16 ldsB[128 * LDT];
  Acc8 acc;
  int m_blk = blockIdx.y * 128, n_blk = blockIdx.x * 128;
  gemm_mainloop(A, B, K, m_blk, n_blk, ldsA, ldsB, acc);
  int lane = threadIdx.x & 31, wave = threadIdx.x >> 5;
  int wm = wave & 3, wn = wave >> 2, half = lane >> 4, l15 = lane & 15;
#pragma unroll
  for (int j = 0; j < 4; ++j) {
    int c = n_blk + wn * 64 + j * 16 + l15;
    float bv = bias[c];
#pragma unroll
    for (int i = 0; i < 2; ++i)
#pragma unroll
      for (int r = 0; r < 8; ++r) {
        int mg = m_blk + wm * 32 + i * 16 + half * 8 + r;
        int b = mg >> 12, q = mg & 4095;
        outWy[((size_t)(b * 256 + c) << 12) + q] = acc.c[i][j][r] + bv;
      }
  }
}

// deterministic per-channel mean/var -> (scale, shift)
__global__ __launch_bounds__(256) void k_bnstats(const float* __restrict__ Wy, const float* __restrict__ gamma,
                                                 const float* __restrict__ beta, float* __restrict__ coefA,
                                                 float* __restrict__ coefB) {
  __shared__ float r1[256], r2[256];
  int c = blockIdx.x, tid = threadIdx.x;
  float s = 0.f, sq = 0.f;
  for (int t = tid; t < MTOT; t += 256) {
    int b = t >> 12, q = t & 4095;
    float v = Wy[((size_t)(b * 256 + c) << 12) + q];
    s += v; sq += v * v;
  }
  r1[tid] = s; r2[tid] = sq;
  __syncthreads();
  for (int st = 128; st > 0; st >>= 1) {
    if (tid < st) { r1[tid] += r1[tid + st]; r2[tid] += r2[tid + st]; }
    __syncthreads();
  }
  if (tid == 0) {
    float inv = 1.f / (float)MTOT;
    float mean = r1[0] * inv;
    float var  = r2[0] * inv - mean * mean;
    float a = gamma[c] * rsqrtf(var + 1e-5f);
    coefA[c] = a;
    coefB[c] = beta[c] - mean * a;
  }
}

// out = Wy*a[c] + b[c] + x   (in-place on d_out)
__global__ __launch_bounds__(256) void k_bnapply(float* __restrict__ out, const float* __restrict__ x,
                                                 const float* __restrict__ coefA, const float* __restrict__ coefB) {
  size_t i = ((size_t)blockIdx.x * 256 + threadIdx.x) * 4;
  int c = (int)((i >> 12) & 255);
  float4 w = *(const float4*)(out + i);
  float4 xv = *(const float4*)(x + i);
  float a = coefA[c], bb = coefB[c];
  w.x = w.x * a + bb + xv.x;
  w.y = w.y * a + bb + xv.y;
  w.z = w.z * a + bb + xv.z;
  w.w = w.w * a + bb + xv.w;
  *(float4*)(out + i) = w;
}

// ---------- host ----------
extern "C" void kernel_launch(void* const* d_in, const int* in_sizes, int n_in,
                              void* d_out, int out_size, void* d_ws, size_t ws_size,
                              hipStream_t stream) {
  const float* x       = (const float*)d_in[0];
  const float* y       = (const float*)d_in[1];
  const float* g_w     = (const float*)d_in[2];
  const float* g_b     = (const float*)d_in[3];
  const float* theta_w = (const float*)d_in[4];
  const float* theta_b = (const float*)d_in[5];
  const float* phi_w   = (const float*)d_in[6];
  const float* phi_b   = (const float*)d_in[7];
  const float* W_w     = (const float*)d_in[8];
  const float* W_b     = (const float*)d_in[9];
  const float* gamma   = (const float*)d_in[10];
  const float* beta    = (const float*)d_in[11];
  float* out = (float*)d_out;

  char* ws = (char*)d_ws;
  size_t off = 0;
  auto alloc = [&](size_t bytes) {
    char* p = ws + off;
    off = (off + bytes + 255) & ~(size_t)255;
    return p;
  };
  u16*   xT     = (u16*)alloc((size_t)MTOT * 256 * 2);
  u16*   yT     = (u16*)alloc((size_t)MTOT * 256 * 2);
  u16*   thw    = (u16*)alloc(128 * 256 * 2);
  u16*   gw     = (u16*)alloc(128 * 256 * 2);
  u16*   phw    = (u16*)alloc(128 * 256 * 2);
  u16*   Www    = (u16*)alloc(256 * 128 * 2);
  u16*   thbf   = (u16*)alloc((size_t)MTOT * 128 * 2);
  float* gconv  = (float*)alloc((size_t)MTOT * 128 * 4);
  float* phconv = (float*)alloc((size_t)MTOT * 128 * 4);
  u16*   gpool  = (u16*)alloc((size_t)BATCH * NK * 128 * 2);
  u16*   phpool = (u16*)alloc((size_t)BATCH * NK * 128 * 2);
  u16*   attn   = (u16*)alloc((size_t)MTOT * 128 * 2);
  float* coefA  = (float*)alloc(256 * 4);
  float* coefB  = (float*)alloc(256 * 4);

  k_transpose_bf16<<<dim3(64, 8, 8), 256, 0, stream>>>(x, xT);
  k_transpose_bf16<<<dim3(64, 8, 8), 256, 0, stream>>>(y, yT);
  k_f32_to_bf16<<<128, 256, 0, stream>>>(theta_w, thw, 128 * 256);
  k_f32_to_bf16<<<128, 256, 0, stream>>>(g_w, gw, 128 * 256);
  k_f32_to_bf16<<<128, 256, 0, stream>>>(phi_w, phw, 128 * 256);
  k_f32_to_bf16<<<128, 256, 0, stream>>>(W_w, Www, 256 * 128);

  k_conv_theta<<<dim3(1, MTOT / 128), 256, 0, stream>>>(xT, thw, theta_b, thbf, 256);
  k_conv_f32<<<dim3(1, MTOT / 128), 256, 0, stream>>>(yT, gw, g_b, gconv, 256, 128);
  k_conv_f32<<<dim3(1, MTOT / 128), 256, 0, stream>>>(yT, phw, phi_b, phconv, 256, 128);
  k_poolnorm<<<BATCH * NK, 128, 0, stream>>>(gconv, phconv, gpool, phpool);
  k_attention<<<dim3(NQ / 128, BATCH), 256, 0, stream>>>(thbf, phpool, gpool, attn);
  k_conv_final<<<dim3(2, MTOT / 128), 256, 0, stream>>>(attn, Www, W_b, out, 128);
  k_bnstats<<<256, 256, 0, stream>>>(out, gamma, beta, coefA, coefB);
  k_bnapply<<<MTOT * 256 / (256 * 4), 256, 0, stream>>>(out, x, coefA, coefB);
}
